// LSTMSensorBasedForwardDynamics_75453985456459
// MI455X (gfx1250) — matrix operand order repaired and checked by reading, not compiled, not run on hardware
//
#include <hip/hip_runtime.h>
#include <hip/hip_bf16.h>

// ---------------- problem constants (from reference) ----------------
#define BATCH   16384
#define TSTEPS  50
#define OBS_DIM 64
#define ACT_DIM 8
#define IN_D    (OBS_DIM + ACT_DIM)   // 72
#define HID     256
#define GATES   (4 * HID)             // 1024
#define KPAD    352                   // 72 + 256 padded to 11*32
#define KTILES  (KPAD / 32)           // 11
#define OUT_DIM 16
#define MROWS   32                    // batch rows per block (2 M-tiles)

typedef __attribute__((ext_vector_type(16))) _Float16 v16h;
typedef __attribute__((ext_vector_type(8)))  _Float16 v8h;
typedef __attribute__((ext_vector_type(8)))  float    v8f;

__device__ __forceinline__ float sigmoidf_(float x) {
    return 1.0f / (1.0f + expf(-x));
}

// -------------------------------------------------------------------
// Pack [W_i ; W_h ; zero-pad] transposed into f16  Wt[n][k], n=0..1023,
// k=0..351 (row stride 352).  B-fragment for wmma_f32_16x16x32_f16 then
// reads 32 contiguous bytes per lane.
// -------------------------------------------------------------------
__global__ void lstm_pack_weights(const float* __restrict__ Wi,
                                  const float* __restrict__ Wh,
                                  _Float16* __restrict__ Wt) {
    int idx = blockIdx.x * blockDim.x + threadIdx.x;
    if (idx >= GATES * KPAD) return;
    int n = idx / KPAD;
    int k = idx - n * KPAD;
    float v = 0.0f;
    if (k < IN_D)            v = Wi[k * GATES + n];
    else if (k < IN_D + HID) v = Wh[(k - IN_D) * GATES + n];
    Wt[n * KPAD + k] = (_Float16)v;
}

// -------------------------------------------------------------------
// Fused LSTM scan (32-row batch tile) + policy heads.
// Block = 256 threads = 8 waves.  Wave w owns hidden cols [w*32,w*32+32)
// across all 4 gates; cell state c lives in registers; each B-fragment
// is shared by two M-tiles (M=32) to halve L2 weight traffic.
// -------------------------------------------------------------------
__global__ __launch_bounds__(256)
void lstm_policy_kernel(const float* __restrict__ obs,
                        const float* __restrict__ act,
                        const _Float16* __restrict__ Wt,
                        const float* __restrict__ bias,
                        const float* __restrict__ muW1, const float* __restrict__ mub1,
                        const float* __restrict__ muW2, const float* __restrict__ mub2,
                        const float* __restrict__ muW3, const float* __restrict__ mub3,
                        const float* __restrict__ lsW1, const float* __restrict__ lsb1,
                        const float* __restrict__ lsW2, const float* __restrict__ lsb2,
                        const float* __restrict__ lsW3, const float* __restrict__ lsb3,
                        float* __restrict__ out) {
    __shared__ __align__(16) _Float16 zsm[MROWS][KPAD]; // [x_t | h | pad] f16
    __shared__ float t1sm[MROWS][64];
    __shared__ float t2sm[MROWS][64];

    const int tid    = threadIdx.x;
    const int wave   = tid >> 5;
    const int lane   = tid & 31;
    const int lanelo = lane & 15;
    const int khalf  = lane >> 4;
    const int r0     = blockIdx.x * MROWS;     // first batch row of this tile

    // ---- init LDS: z h-region + pad = 0 ----
    for (int idx = tid; idx < MROWS * (KPAD - IN_D); idx += 256) {
        int row = idx / (KPAD - IN_D);
        int k   = idx % (KPAD - IN_D);
        zsm[row][IN_D + k] = (_Float16)0.0f;
    }
    __syncthreads();

    // ---- per-lane gate biases (loop-invariant over t) ----
    float bi[2], bf[2], bg[2], bo[2];
#pragma unroll
    for (int s = 0; s < 2; ++s) {
        int j = wave * 32 + s * 16 + lanelo;
        bi[s] = bias[0 * HID + j];
        bf[s] = bias[1 * HID + j];
        bg[s] = bias[2 * HID + j];
        bo[s] = bias[3 * HID + j];
    }

    // ---- cell state in registers: creg[mtile][s][v] ----
    float creg[2][2][8];
#pragma unroll
    for (int m = 0; m < 2; ++m)
#pragma unroll
        for (int s = 0; s < 2; ++s)
#pragma unroll
            for (int v = 0; v < 8; ++v)
                creg[m][s][v] = 0.0f;

    const v8f vzero = {0.f, 0.f, 0.f, 0.f, 0.f, 0.f, 0.f, 0.f};

    // =================== sequential scan over T ===================
    for (int t = 0; t < TSTEPS; ++t) {
        // Launder a 64-bit *offset* (not the pointer) so the weight
        // fragment loads cannot be hoisted/CSE'd out of the t-loop
        // (avoids the 700-VGPR spill-to-scratch), while the pointer
        // provenance survives -> global_load_b128, not flat_load.
        size_t wofs = 0;
        asm volatile("" : "+s"(wofs));
        const _Float16* wp = Wt + wofs;

        // ---- stage x_t (f32 -> f16) into z[:, 0:72] ----
        for (int idx = tid; idx < MROWS * IN_D; idx += 256) {
            int row = idx / IN_D;
            int k   = idx - row * IN_D;
            float v;
            if (k < OBS_DIM) {
                const float* p = &obs[(size_t)(r0 + row) * TSTEPS * OBS_DIM + t * OBS_DIM + k];
                v = *p;
                if (t + 1 < TSTEPS)                        // warm next frame (L2)
                    __builtin_prefetch(p + OBS_DIM, 0, 1); // -> global_prefetch_b8
            } else {
                v = act[(size_t)(r0 + row) * TSTEPS * ACT_DIM + t * ACT_DIM + (k - OBS_DIM)];
            }
            zsm[row][k] = (_Float16)v;
        }
        __syncthreads();

        // ---- gates[32 x 1024] = z @ [Wi;Wh]  via f16 WMMA, f32 acc ----
        v8f acc[2][4][2];
#pragma unroll
        for (int m = 0; m < 2; ++m)
#pragma unroll
            for (int g = 0; g < 4; ++g)
#pragma unroll
                for (int s = 0; s < 2; ++s)
                    acc[m][g][s] = vzero;

        for (int kt = 0; kt < KTILES; ++kt) {
            // A fragments (16x32 f16) from LDS, in contiguous k order (k = 16*half + s), not the ISA table order:
            // lanes 0-15: M=lane, K = 0..15; lanes 16-31: K = 16..31 (matches B)
            v16h afrag[2];
#pragma unroll
            for (int m = 0; m < 2; ++m) {
                union { v16h v; v8h h[2]; } ua;
                const _Float16* ap = &zsm[m * 16 + lanelo][kt * 32 + khalf * 16];
                ua.h[0] = *(const v8h*)(ap);
                ua.h[1] = *(const v8h*)(ap + 8);
                afrag[m] = ua.v;
            }

#pragma unroll
            for (int g = 0; g < 4; ++g) {
#pragma unroll
                for (int s = 0; s < 2; ++s) {
                    const int ntile = g * 16 + wave * 2 + s;
                    // B fragment (32x16 f16): lane n, 16 consecutive K values
                    const _Float16* bp =
                        wp + (size_t)(ntile * 16 + lanelo) * KPAD + kt * 32 + khalf * 16;
                    const v16h bfrag = *(const v16h*)bp;
#pragma unroll
                    for (int m = 0; m < 2; ++m) {
                        acc[m][g][s] = __builtin_amdgcn_wmma_f32_16x16x32_f16(
                            false, afrag[m], false, bfrag, (short)0,
                            acc[m][g][s], false, false);
                    }
                }
            }
        }
        __syncthreads();   // all waves done reading z (previous h)

        // ---- cell update entirely in registers ----
#pragma unroll
        for (int m = 0; m < 2; ++m) {
#pragma unroll
            for (int s = 0; s < 2; ++s) {
                const int j = wave * 32 + s * 16 + lanelo;
#pragma unroll
                for (int v = 0; v < 8; ++v) {
                    const int row = m * 16 + khalf * 8 + v;
                    float iv = acc[m][0][s][v] + bi[s];
                    float fv = acc[m][1][s][v] + bf[s];
                    float gv = acc[m][2][s][v] + bg[s];
                    float ov = acc[m][3][s][v] + bo[s];
                    float nc = sigmoidf_(fv) * creg[m][s][v]
                             + sigmoidf_(iv) * tanhf(gv);
                    float nh = sigmoidf_(ov) * tanhf(nc);
                    creg[m][s][v] = nc;
                    zsm[row][IN_D + j] = (_Float16)nh;  // feed next step + heads
                }
            }
        }
        __syncthreads();
    }

    // =================== heads on final h (tiny, VALU) ===================
    // h is read back from the f16 staging buffer.
    // mu: 256 -> 64 -> 64 -> 16
    for (int idx = tid; idx < MROWS * 64; idx += 256) {
        int row = idx >> 6, col = idx & 63;
        float a = mub1[col];
#pragma unroll 4
        for (int k = 0; k < HID; ++k)
            a += (float)zsm[row][IN_D + k] * muW1[k * 64 + col];
        t1sm[row][col] = fmaxf(a, 0.0f);
    }
    __syncthreads();
    for (int idx = tid; idx < MROWS * 64; idx += 256) {
        int row = idx >> 6, col = idx & 63;
        float a = mub2[col];
#pragma unroll 4
        for (int k = 0; k < 64; ++k) a += t1sm[row][k] * muW2[k * 64 + col];
        t2sm[row][col] = fmaxf(a, 0.0f);
    }
    __syncthreads();
    for (int idx = tid; idx < MROWS * OUT_DIM; idx += 256) {
        int row = idx >> 4, col = idx & 15;
        float a = mub3[col];
#pragma unroll 4
        for (int k = 0; k < 64; ++k) a += t2sm[row][k] * muW3[k * OUT_DIM + col];
        out[(size_t)(r0 + row) * OUT_DIM + col] = a;
    }
    __syncthreads();

    // log_std: 256 -> 32 -> 32 -> 16, clip, exp
    for (int idx = tid; idx < MROWS * 32; idx += 256) {
        int row = idx >> 5, col = idx & 31;
        float a = lsb1[col];
#pragma unroll 4
        for (int k = 0; k < HID; ++k)
            a += (float)zsm[row][IN_D + k] * lsW1[k * 32 + col];
        t1sm[row][col] = fmaxf(a, 0.0f);
    }
    __syncthreads();
    for (int idx = tid; idx < MROWS * 32; idx += 256) {
        int row = idx >> 5, col = idx & 31;
        float a = lsb2[col];
#pragma unroll 4
        for (int k = 0; k < 32; ++k) a += t1sm[row][k] * lsW2[k * 32 + col];
        t2sm[row][col] = fmaxf(a, 0.0f);
    }
    __syncthreads();
    for (int idx = tid; idx < MROWS * OUT_DIM; idx += 256) {
        int row = idx >> 4, col = idx & 15;
        float a = lsb3[col];
#pragma unroll 4
        for (int k = 0; k < 32; ++k) a += t2sm[row][k] * lsW3[k * OUT_DIM + col];
        a = fminf(fmaxf(a, -10.0f), 2.0f);
        out[(size_t)BATCH * OUT_DIM + (size_t)(r0 + row) * OUT_DIM + col] = expf(a);
    }
}

// -------------------------------------------------------------------
extern "C" void kernel_launch(void* const* d_in, const int* in_sizes, int n_in,
                              void* d_out, int out_size, void* d_ws, size_t ws_size,
                              hipStream_t stream) {
    (void)in_sizes; (void)n_in; (void)out_size; (void)ws_size;

    const float* obs  = (const float*)d_in[0];
    const float* actp = (const float*)d_in[1];
    const float* Wi   = (const float*)d_in[2];
    const float* Wh   = (const float*)d_in[3];
    const float* bias = (const float*)d_in[4];
    const float* muW1 = (const float*)d_in[5];
    const float* mub1 = (const float*)d_in[6];
    const float* muW2 = (const float*)d_in[7];
    const float* mub2 = (const float*)d_in[8];
    const float* muW3 = (const float*)d_in[9];
    const float* mub3 = (const float*)d_in[10];
    const float* lsW1 = (const float*)d_in[11];
    const float* lsb1 = (const float*)d_in[12];
    const float* lsW2 = (const float*)d_in[13];
    const float* lsb2 = (const float*)d_in[14];
    const float* lsW3 = (const float*)d_in[15];
    const float* lsb3 = (const float*)d_in[16];

    _Float16* Wt = (_Float16*)d_ws;   // 1024 * 352 * 2B = 704 KB

    {
        int total = GATES * KPAD;
        lstm_pack_weights<<<(total + 255) / 256, 256, 0, stream>>>(Wi, Wh, Wt);
    }
    lstm_policy_kernel<<<BATCH / MROWS, 256, 0, stream>>>(
        obs, actp, Wt, bias,
        muW1, mub1, muW2, mub2, muW3, mub3,
        lsW1, lsb1, lsW2, lsb2, lsW3, lsb3,
        (float*)d_out);
}
